// PVB_11673721111014
// MI455X (gfx1250) — matrix operand order repaired and checked by reading, not compiled, not run on hardware
//
#include <hip/hip_runtime.h>

typedef __attribute__((ext_vector_type(16))) _Float16 v16h;
typedef __attribute__((ext_vector_type(8)))  float    v8f;

constexpr int B_ = 4, CH_ = 64, N_ = 16384, K_ = 16, R_ = 16, R3_ = R_ * R_ * R_;

__device__ __forceinline__ float bn1(float x, float g, float b, float m, float v, float eps) {
  return (x - m) * (g * rsqrtf(v + eps)) + b;
}

// ---------------- zero scratch ----------------
__global__ __launch_bounds__(256) void k_zero(float* __restrict__ p, int n) {
  int i = blockIdx.x * 256 + threadIdx.x;
  if (i < n) p[i] = 0.0f;
}

// ---------------- per (b,dim) mean of xyz ----------------
__global__ __launch_bounds__(256) void k_mean(const float* __restrict__ xyz, float* __restrict__ mean) {
  __shared__ float red[256];
  int bd = blockIdx.x;                       // [0, 3*B)
  float s = 0.0f;
  for (int i = threadIdx.x; i < N_; i += 256) s += xyz[bd * N_ + i];
  red[threadIdx.x] = s;
  __syncthreads();
  for (int off = 128; off > 0; off >>= 1) {
    if (threadIdx.x < off) red[threadIdx.x] += red[threadIdx.x + off];
    __syncthreads();
  }
  if (threadIdx.x == 0) mean[bd] = red[0] / (float)N_;
}

// ---------------- voxelize scatter (wave per point) ----------------
__global__ __launch_bounds__(256) void k_vox(const float* __restrict__ x, const float* __restrict__ xyz,
                                             const float* __restrict__ mean, float* __restrict__ ncw,
                                             float* __restrict__ cnt, float* __restrict__ sums) {
  int lane = threadIdx.x & 31, wave = threadIdx.x >> 5;
  int p = blockIdx.x * 8 + wave;
  int b = p >> 14, n = p & (N_ - 1);
  float nc[3];
#pragma unroll
  for (int d = 0; d < 3; ++d) {
    float v = xyz[(b * 3 + d) * N_ + n] - mean[b * 3 + d];
    v = (v + 1.0f) * 0.5f * (float)R_;
    v = fminf(fmaxf(v, 0.0f), (float)(R_ - 1));
    nc[d] = v;
    if (lane == 0) ncw[(b * 3 + d) * N_ + n] = v;
  }
  int vi0 = (int)rintf(nc[0]), vi1 = (int)rintf(nc[1]), vi2 = (int)rintf(nc[2]);
  int flat = (vi0 * R_ + vi1) * R_ + vi2;
  float* sb = sums + (size_t)(b * R3_ + flat) * CH_;
  atomicAdd(sb + lane,      x[(b * CH_ + lane) * N_ + n]);
  atomicAdd(sb + lane + 32, x[(b * CH_ + lane + 32) * N_ + n]);
  if (lane == 0) atomicAdd(cnt + b * R3_ + flat, 1.0f);
}

// ---------------- voxel mean ----------------
__global__ __launch_bounds__(256) void k_norm(const float* __restrict__ sums, const float* __restrict__ cnt,
                                              float* __restrict__ v0, int total) {
  int i = blockIdx.x * 256 + threadIdx.x;
  if (i < total) v0[i] = sums[i] / fmaxf(cnt[i / CH_], 1.0f);
}

// ---------------- depthwise 3x3x3 conv + BN + LeakyReLU ----------------
__global__ __launch_bounds__(256) void k_conv(const float* __restrict__ vin, const float* __restrict__ w,
                                              const float* __restrict__ bias, const float* __restrict__ bn,
                                              float* __restrict__ vout, float eps) {
  int t = blockIdx.x * 256 + threadIdx.x;    // [0, B*R3*CH)
  int c = t & (CH_ - 1);
  int vox = (t >> 6) & (R3_ - 1);
  int b = t >> 18;
  int X = vox >> 8, Y = (vox >> 4) & 15, Z = vox & 15;
  float acc = bias[c];
  const float* wc = w + c * 27;
#pragma unroll
  for (int dx = 0; dx < 3; ++dx) {
    int xx = X + dx - 1;
    if (xx < 0 || xx >= R_) continue;
#pragma unroll
    for (int dy = 0; dy < 3; ++dy) {
      int yy = Y + dy - 1;
      if (yy < 0 || yy >= R_) continue;
#pragma unroll
      for (int dz = 0; dz < 3; ++dz) {
        int zz = Z + dz - 1;
        if (zz < 0 || zz >= R_) continue;
        acc += wc[(dx * 3 + dy) * 3 + dz] * vin[(size_t)(b * R3_ + ((xx * R_ + yy) * R_ + zz)) * CH_ + c];
      }
    }
  }
  float y = bn1(acc, bn[c], bn[CH_ + c], bn[2 * CH_ + c], bn[3 * CH_ + c], eps);
  vout[t] = (y < 0.0f) ? 0.1f * y : y;
}

// ---------------- fused devoxelize + WMMA projection + neighbor max ----------------
__global__ __launch_bounds__(256) void k_final(
    const float* __restrict__ x, const float* __restrict__ xyz, const int* __restrict__ idx,
    const float* __restrict__ v2, const float* __restrict__ ncw,
    const float* __restrict__ pw1, const float* __restrict__ pbn,
    const float* __restrict__ pw2, const float* __restrict__ ptbn,
    float* __restrict__ out) {
  __shared__ _Float16 dfT[8][K_][CH_];   // per-wave 64x16 f16 df tile (16 KB)
  __shared__ float    dpT[8][K_][6];     // per-wave dp weights (3 KB)
  int lane = threadIdx.x & 31, wave = threadIdx.x >> 5;
  int p = blockIdx.x * 8 + wave;
  int b = p >> 14, n = p & (N_ - 1);
  int c0 = lane, c1 = lane + 32;

  // A operand: proj_w1 [3x64] zero-padded to 16x32, split into two K-chunks.
  // 16-bit A 16x32 layout: lanes 0-15 -> K {0..7,16..23}; lanes 16-31 -> K {8..15,24..31}; M = lane&15.
  v16h a0, a1;
  {
    int M = lane & 15;
    int hi = lane >> 4;
#pragma unroll
    for (int j = 0; j < 16; ++j) {
      int kk = ((j < 8) ? j : (j + 8)) + hi * 8;
      float w0 = 0.0f, w1 = 0.0f;
      if (M < 3) { w0 = pw1[M * CH_ + kk]; w1 = pw1[M * CH_ + 32 + kk]; }
      a0[j] = (_Float16)w0;
      a1[j] = (_Float16)w1;
    }
  }

  // ----- trilinear devoxelize (lane covers channels c0, c1) -----
  float nc0 = ncw[(b * 3 + 0) * N_ + n], nc1 = ncw[(b * 3 + 1) * N_ + n], nc2 = ncw[(b * 3 + 2) * N_ + n];
  int lo0 = (int)floorf(nc0), lo1 = (int)floorf(nc1), lo2 = (int)floorf(nc2);
  float f0 = nc0 - lo0, f1 = nc1 - lo1, f2 = nc2 - lo2;
  int h0i = min(lo0 + 1, R_ - 1), h1i = min(lo1 + 1, R_ - 1), h2i = min(lo2 + 1, R_ - 1);
  float pv0 = 0.0f, pv1 = 0.0f;
#pragma unroll
  for (int corner = 0; corner < 8; ++corner) {
    int dx = corner >> 2, dy = (corner >> 1) & 1, dz = corner & 1;
    int xi = dx ? h0i : lo0, yi = dy ? h1i : lo1, zi = dz ? h2i : lo2;
    float wgt = (dx ? f0 : 1.0f - f0) * (dy ? f1 : 1.0f - f1) * (dz ? f2 : 1.0f - f2);
    const float* vb = v2 + (size_t)(b * R3_ + (xi * R_ + yi) * R_ + zi) * CH_;
    pv0 += wgt * vb[c0];
    pv1 += wgt * vb[c1];
  }

  // ----- gather neighbors; build df tile in LDS -----
  float xc0 = x[(b * CH_ + c0) * N_ + n], xc1 = x[(b * CH_ + c1) * N_ + n];
  float xj0[K_], xj1[K_];
  const int* ib = idx + (size_t)(b * N_ + n) * K_;
#pragma unroll
  for (int k = 0; k < K_; ++k) {
    int j = ib[k];
    float u0 = x[(b * CH_ + c0) * N_ + j], u1 = x[(b * CH_ + c1) * N_ + j];
    xj0[k] = u0; xj1[k] = u1;
    dfT[wave][k][c0] = (_Float16)(u0 - xc0);
    dfT[wave][k][c1] = (_Float16)(u1 - xc1);
  }
  __builtin_amdgcn_wave_barrier();   // keep LDS stores before tile reads

  // B operand: 16-bit B 32x16 layout: column N = lane&15; lanes 0-15 K {0..7,16..23}, lanes 16-31 K {8..15,24..31}.
  int col = lane & 15;
  int cb = (lane >> 4) * 8;
  v16h b0, b1;
#pragma unroll
  for (int j = 0; j < 16; ++j) {
    b0[j] = dfT[wave][col][cb + ((j < 8) ? j : (j + 8))];        // channels chunk 0
    b1[j] = dfT[wave][col][32 + cb + ((j < 8) ? j : (j + 8))];   // channels chunk 1
  }

  // h[16x16] = pw1_pad[16x32] * df0[32x16] + pw1_pad[16x32] * df1[32x16]
  v8f h = {};
  h = __builtin_amdgcn_wmma_f32_16x16x32_f16(false, a0, false, b0, (short)0, h, false, false);
  h = __builtin_amdgcn_wmma_f32_16x16x32_f16(false, a1, false, b1, (short)0, h, false, false);

  // lanes 0..15 hold rows M=0..7 -> D regs 0..2 are h[0..2][k=lane]
  if (lane < 16) {
    int k = lane;
    float hb[3];
#pragma unroll
    for (int o = 0; o < 3; ++o) {
      float t = bn1(h[o], pbn[o], pbn[3 + o], pbn[6 + o], pbn[9 + o], 1e-5f);
      hb[o] = fmaxf(t, 0.0f);
    }
    int j = ib[k];
#pragma unroll
    for (int o = 0; o < 3; ++o) {
      dpT[wave][k][o]     = xyz[(b * 3 + o) * N_ + j] - xyz[(b * 3 + o) * N_ + n];
      dpT[wave][k][3 + o] = pw2[o * 3 + 0] * hb[0] + pw2[o * 3 + 1] * hb[1] + pw2[o * 3 + 2] * hb[2];
    }
  }
  __builtin_amdgcn_wave_barrier();

  // ----- channel-group weighting, pt_bn + ReLU, max over K -----
  int g0 = min(c0 / 10, 5), g1 = min(c1 / 10, 5);
  float s0 = ptbn[c0] * rsqrtf(ptbn[3 * CH_ + c0] + 1e-5f);
  float s1 = ptbn[c1] * rsqrtf(ptbn[3 * CH_ + c1] + 1e-5f);
  float be0 = ptbn[CH_ + c0], be1 = ptbn[CH_ + c1];
  float me0 = ptbn[2 * CH_ + c0], me1 = ptbn[2 * CH_ + c1];
  float mx0 = 0.0f, mx1 = 0.0f;
#pragma unroll
  for (int k = 0; k < K_; ++k) {
    float w0 = dpT[wave][k][g0], w1 = dpT[wave][k][g1];
    float t0 = (xj0[k] * w0 - me0) * s0 + be0;
    float t1 = (xj1[k] * w1 - me1) * s1 + be1;
    mx0 = fmaxf(mx0, fmaxf(t0, 0.0f));   // ReLU then max (max of nonneg == init 0)
    mx1 = fmaxf(mx1, fmaxf(t1, 0.0f));
  }
  out[(b * CH_ + c0) * N_ + n] = pv0 + mx0;
  out[(b * CH_ + c1) * N_ + n] = pv1 + mx1;
}

extern "C" void kernel_launch(void* const* d_in, const int* in_sizes, int n_in,
                              void* d_out, int out_size, void* d_ws, size_t ws_size,
                              hipStream_t stream) {
  (void)in_sizes; (void)n_in; (void)out_size; (void)ws_size;
  const float* x     = (const float*)d_in[0];
  const float* xyz   = (const float*)d_in[1];
  const int*   idx   = (const int*)d_in[2];
  const float* w3d1  = (const float*)d_in[3];
  const float* b3d1  = (const float*)d_in[4];
  const float* bn3d1 = (const float*)d_in[5];
  const float* w3d2  = (const float*)d_in[6];
  const float* b3d2  = (const float*)d_in[7];
  const float* bn3d2 = (const float*)d_in[8];
  const float* pw1   = (const float*)d_in[9];
  const float* pbn   = (const float*)d_in[10];
  const float* pw2   = (const float*)d_in[11];
  const float* ptbn  = (const float*)d_in[12];
  float* out = (float*)d_out;

  // workspace layout (floats): mean | nc | cnt | sums | v0 | v1 | v2
  float* ws   = (float*)d_ws;
  float* mean = ws;                        // 12 (padded to 16)
  float* ncw  = ws + 16;                   // 3*B*N
  float* cnt  = ncw + 3 * B_ * N_;         // B*R3
  float* sums = cnt + B_ * R3_;            // B*R3*CH  (contiguous with cnt)
  float* v0   = sums + B_ * R3_ * CH_;
  float* v1   = v0 + B_ * R3_ * CH_;
  float* v2   = v1 + B_ * R3_ * CH_;

  int zeroN = B_ * R3_ + B_ * R3_ * CH_;   // cnt + sums
  hipLaunchKernelGGL(k_zero, dim3((zeroN + 255) / 256), dim3(256), 0, stream, cnt, zeroN);
  hipLaunchKernelGGL(k_mean, dim3(3 * B_), dim3(256), 0, stream, xyz, mean);
  hipLaunchKernelGGL(k_vox, dim3(B_ * N_ / 8), dim3(256), 0, stream, x, xyz, mean, ncw, cnt, sums);
  int tot = B_ * R3_ * CH_;
  hipLaunchKernelGGL(k_norm, dim3(tot / 256), dim3(256), 0, stream, sums, cnt, v0, tot);
  hipLaunchKernelGGL(k_conv, dim3(tot / 256), dim3(256), 0, stream, v0, w3d1, b3d1, bn3d1, v1, 1e-4f);
  hipLaunchKernelGGL(k_conv, dim3(tot / 256), dim3(256), 0, stream, v1, w3d2, b3d2, bn3d2, v2, 1e-4f);
  hipLaunchKernelGGL(k_final, dim3(B_ * N_ / 8), dim3(256), 0, stream,
                     x, xyz, idx, v2, ncw, pw1, pbn, pw2, ptbn, out);
}
